// switch_NGP_61667140436310
// MI455X (gfx1250) — compile-verified, matrix operand order checked by reading, not run on hardware
//
#include <hip/hip_runtime.h>
#include <hip/hip_bf16.h>
#include <math.h>

typedef __attribute__((ext_vector_type(16))) _Float16 v16h;
typedef __attribute__((ext_vector_type(8)))  _Float16 v8h;
typedef __attribute__((ext_vector_type(8)))  float    v8f;

#define N_PTS   262144
#define NLEV    16
#define TBL     524288u
#define TMASK   524287u
#define WPB     4            // waves per block
#define TPB     128          // threads per block
#define NBLK    (N_PTS / (WPB * 16))   // 4096

// ---- LDS arena offsets (in halfs) ----
#define W_GATE0 0            // [64][32]
#define W_GATE1 2048         // [64][64]
#define W_GATE2 6144         // [16][64]  (out padded 2->16)
#define W_EXP0  7168         // [2][64][32]
#define W_EXP1  11264        // [2][64][64]
#define W_EXP2  19456        // [2][32][64]
#define W_GEO0  23552        // [64][32]
#define W_GEO1  25600        // [32][64]  (out padded 17->32)
#define W_RGB0  27648        // [64][32]
#define W_RGB1  29696        // [64][64]
#define W_RGB2  33792        // [16][64]  (out padded 3->16)
#define W_TOTAL 34816
// per-wave region (halfs)
#define PW_A    0            // [16][64] f16 act
#define PW_B    1024
#define PW_C    2048
#define PW_SCR  3072         // 512 floats (as 1024 halfs)
#define PW_SEL  4096         // 16 floats (as 32 halfs)
#define PER_WAVE 4128
#define SMEM_HALFS (W_TOTAL + WPB * PER_WAVE)

extern __shared__ _Float16 smemH[];

// single-instruction ReLU: v_med3_f32(v, 0, +inf)
__device__ __forceinline__ float reluf(float v) {
  return __builtin_amdgcn_fmed3f(v, 0.0f, __builtin_inff());
}

// Load one 16x32 f16 fragment (A or B) from row-major [row][K] LDS.
// rowPtr must point at element [row][kbase].
__device__ __forceinline__ v16h ldFragRow(const _Float16* rowPtr, int lane) {
  const int kh = (lane >> 4) << 3;                 // 0 or 8 halfs
  v8h lo = *(const v8h*)(rowPtr + kh);
  v8h hi = *(const v8h*)(rowPtr + kh + 16);
  v16h f;
#pragma unroll
  for (int i = 0; i < 8; ++i) { f[i] = lo[i]; f[i + 8] = hi[i]; }
  return f;
}

// act[16][K] (f16, LDS) @ wT[NOUT][K] (f16, LDS) -> out[16][NOUT] (f16, LDS)
template<int K, int NOUT, bool RELU>
__device__ __forceinline__ void mlpLayer(const _Float16* inAct, const _Float16* wT,
                                         _Float16* outAct, int lane) {
  const int row = lane & 15;
  v16h a[K / 32];
#pragma unroll
  for (int kb = 0; kb < K / 32; ++kb)
    a[kb] = ldFragRow(inAct + row * K + kb * 32, lane);
  const int col = lane & 15;
  const int mb  = (lane >> 4) << 3;
#pragma unroll
  for (int nt = 0; nt < NOUT / 16; ++nt) {
    v8f acc = {};
#pragma unroll
    for (int kb = 0; kb < K / 32; ++kb) {
      v16h b = ldFragRow(wT + (nt * 16 + row) * K + kb * 32, lane);
      acc = __builtin_amdgcn_wmma_f32_16x16x32_f16(false, a[kb], false, b,
                                                   (short)0, acc, false, false);
    }
#pragma unroll
    for (int r = 0; r < 8; ++r) {
      float v = acc[r];
      if (RELU) v = reluf(v);
      outAct[(mb + r) * NOUT + nt * 16 + col] = (_Float16)v;
    }
  }
}

// Same but final layer: store f32 tiles to scratch [16][NOUT].
template<int K, int NOUT>
__device__ __forceinline__ void mlpLayerF32(const _Float16* inAct, const _Float16* wT,
                                            float* outScr, int lane) {
  const int row = lane & 15;
  v16h a[K / 32];
#pragma unroll
  for (int kb = 0; kb < K / 32; ++kb)
    a[kb] = ldFragRow(inAct + row * K + kb * 32, lane);
  const int col = lane & 15;
  const int mb  = (lane >> 4) << 3;
#pragma unroll
  for (int nt = 0; nt < NOUT / 16; ++nt) {
    v8f acc = {};
#pragma unroll
    for (int kb = 0; kb < K / 32; ++kb) {
      v16h b = ldFragRow(wT + (nt * 16 + row) * K + kb * 32, lane);
      acc = __builtin_amdgcn_wmma_f32_16x16x32_f16(false, a[kb], false, b,
                                                   (short)0, acc, false, false);
    }
#pragma unroll
    for (int r = 0; r < 8; ++r)
      outScr[(mb + r) * NOUT + nt * 16 + col] = acc[r];
  }
}

// transpose-convert f32 weight [K][N] (row-major) -> f16 [n][k] in LDS
__device__ __forceinline__ void fillT(const float* src, int K, int N,
                                      _Float16* dst, int tid) {
  for (int i = tid; i < K * N; i += TPB) {
    int k = i / N, n = i % N;
    dst[n * K + k] = (_Float16)src[i];
  }
}

__global__ void __launch_bounds__(TPB)
ngp_moe_kernel(const float* __restrict__ X, const float* __restrict__ D,
               const float* __restrict__ HT,
               const float* gw0, const float* gw1, const float* gw2,
               const float* ew0, const float* ew1, const float* ew2,
               const float* geow0, const float* geow1,
               const float* rw0, const float* rw1, const float* rw2,
               float* __restrict__ out) {
  const int tid    = threadIdx.x;
  const int waveId = tid >> 5;
  const int lane   = tid & 31;

  float* outSigma = out;
  float* outRgb   = out + N_PTS;
  float* outGates = out + 4 * N_PTS;
  float* outLoad  = out + 6 * N_PTS;
  float* outTop   = out + 6 * N_PTS + 2;

  // ---------------- stage weights (f16, transposed) into LDS ----------------
  for (int i = tid; i < W_TOTAL; i += TPB) smemH[i] = (_Float16)0.0f;
  __syncthreads();
  fillT(gw0, 32, 64, smemH + W_GATE0, tid);
  fillT(gw1, 64, 64, smemH + W_GATE1, tid);
  fillT(gw2, 64, 2,  smemH + W_GATE2, tid);
  fillT(ew0,            32, 64, smemH + W_EXP0,        tid);
  fillT(ew0 + 32 * 64,  32, 64, smemH + W_EXP0 + 2048, tid);
  fillT(ew1,            64, 64, smemH + W_EXP1,        tid);
  fillT(ew1 + 64 * 64,  64, 64, smemH + W_EXP1 + 4096, tid);
  fillT(ew2,            64, 32, smemH + W_EXP2,        tid);
  fillT(ew2 + 64 * 32,  64, 32, smemH + W_EXP2 + 2048, tid);
  fillT(geow0, 32, 64, smemH + W_GEO0, tid);
  fillT(geow1, 64, 17, smemH + W_GEO1, tid);
  fillT(rw0, 32, 64, smemH + W_RGB0, tid);
  fillT(rw1, 64, 64, smemH + W_RGB1, tid);
  fillT(rw2, 64, 3,  smemH + W_RGB2, tid);
  __syncthreads();

  _Float16* base = smemH + W_TOTAL + waveId * PER_WAVE;
  _Float16* bufA = base + PW_A;
  _Float16* bufB = base + PW_B;
  _Float16* bufC = base + PW_C;
  float*    scrF = (float*)(base + PW_SCR);
  float*    selW = (float*)(base + PW_SEL);

  const int tileBase = (blockIdx.x * WPB + waveId) * 16;

  // ---------------- 1. hash encoding -> bufA [16 pts][32 feats] -------------
#pragma unroll 2
  for (int it = 0; it < 8; ++it) {
    int task = it * 32 + lane;          // 256 (point,level) tasks
    int p    = task & 15;
    int lvl  = task >> 4;
    int gp   = tileBase + p;
    float x0 = X[gp * 3 + 0], x1 = X[gp * 3 + 1], x2 = X[gp * 3 + 2];
    x0 = fminf(fmaxf((x0 + 1.0f) * 0.5f, 0.0f), 1.0f);
    x1 = fminf(fmaxf((x1 + 1.0f) * 0.5f, 0.0f), 1.0f);
    x2 = fminf(fmaxf((x2 + 1.0f) * 0.5f, 0.0f), 1.0f);
    float sc = 16.0f * exp2f((float)lvl * (7.0f / 15.0f)) - 1.0f;
    float p0 = x0 * sc + 0.5f, p1 = x1 * sc + 0.5f, p2 = x2 * sc + 0.5f;
    float f0 = floorf(p0), f1 = floorf(p1), f2 = floorf(p2);
    float r0 = p0 - f0, r1 = p1 - f1, r2 = p2 - f2;
    unsigned i0 = (unsigned)f0, i1 = (unsigned)f1, i2 = (unsigned)f2;
    float a0 = 0.0f, a1 = 0.0f;
#pragma unroll
    for (int c = 0; c < 8; ++c) {
      unsigned cx = i0 + ((c >> 2) & 1);
      unsigned cy = i1 + ((c >> 1) & 1);
      unsigned cz = i2 + (c & 1);
      unsigned h  = cx ^ (cy * 2654435761u) ^ (cz * 805459861u);
      unsigned idx = h & TMASK;
      const float* fp = HT + ((size_t)lvl * TBL + idx) * 2;
      float w = ((c & 4) ? r0 : 1.0f - r0) *
                ((c & 2) ? r1 : 1.0f - r1) *
                ((c & 1) ? r2 : 1.0f - r2);
      a0 += w * fp[0];
      a1 += w * fp[1];
    }
    bufA[p * 32 + 2 * lvl + 0] = (_Float16)a0;
    bufA[p * 32 + 2 * lvl + 1] = (_Float16)a1;
  }
  __syncthreads();

  // ---------------- 2. gate MLP + top-1 select ------------------------------
  mlpLayer<32, 64, true>(bufA, smemH + W_GATE0, bufB, lane);
  __syncthreads();
  mlpLayer<64, 64, true>(bufB, smemH + W_GATE1, bufC, lane);
  __syncthreads();
  mlpLayerF32<64, 16>(bufC, smemH + W_GATE2, scrF, lane);
  __syncthreads();
  {
    int p = lane & 15;
    float l0 = scrF[p * 16 + 0], l1 = scrF[p * 16 + 1];
    bool top = l1 > l0;
    if (lane < 16) {
      int gp = tileBase + p;
      outGates[gp * 2 + 0] = top ? 0.0f : 1.0f;
      outGates[gp * 2 + 1] = top ? 1.0f : 0.0f;
      outTop[gp]           = top ? 1.0f : 0.0f;
      selW[p]              = top ? 1.0f : 0.0f;
    }
    unsigned long long m = __ballot(top && (lane < 16));
    int c1 = __popcll(m);
    if (lane == 0) {
      atomicAdd(&outLoad[0], (float)(16 - c1));
      atomicAdd(&outLoad[1], (float)c1);
    }
  }
  __syncthreads();

  // ---------------- 3. both experts, masked combine -> bufB [16][32] --------
  v8f eo[2][2];
#pragma unroll
  for (int e = 0; e < 2; ++e) {
    mlpLayer<32, 64, true>(bufA, smemH + W_EXP0 + e * 2048, bufB, lane);
    __syncthreads();
    mlpLayer<64, 64, true>(bufB, smemH + W_EXP1 + e * 4096, bufC, lane);
    __syncthreads();
    { // last layer 64->32, keep f32 tiles in regs
      const int row = lane & 15;
      v16h a0 = ldFragRow(bufC + row * 64 + 0,  lane);
      v16h a1 = ldFragRow(bufC + row * 64 + 32, lane);
      const _Float16* wT = smemH + W_EXP2 + e * 2048;
#pragma unroll
      for (int nt = 0; nt < 2; ++nt) {
        v8f acc = {};
        v16h b0 = ldFragRow(wT + (nt * 16 + row) * 64 + 0,  lane);
        acc = __builtin_amdgcn_wmma_f32_16x16x32_f16(false, a0, false, b0,
                                                     (short)0, acc, false, false);
        v16h b1 = ldFragRow(wT + (nt * 16 + row) * 64 + 32, lane);
        acc = __builtin_amdgcn_wmma_f32_16x16x32_f16(false, a1, false, b1,
                                                     (short)0, acc, false, false);
        eo[e][nt] = acc;
      }
    }
    __syncthreads();
  }
  {
    const int col = lane & 15;
    const int mb  = (lane >> 4) << 3;
#pragma unroll
    for (int nt = 0; nt < 2; ++nt)
#pragma unroll
      for (int r = 0; r < 8; ++r) {
        int m = mb + r;
        float v = (selW[m] > 0.5f) ? eo[1][nt][r] : eo[0][nt][r];
        bufB[m * 32 + nt * 16 + col] = (_Float16)v;
      }
  }
  __syncthreads();

  // ---------------- 4. geo MLP: 32->64 relu, 64->17 -------------------------
  mlpLayer<32, 64, true>(bufB, smemH + W_GEO0, bufC, lane);
  __syncthreads();
  mlpLayerF32<64, 32>(bufC, smemH + W_GEO1, scrF, lane);   // [16][32], cols 0..16 valid
  __syncthreads();

  // sigma + build rgb input in bufA: cols 0..15 = SH(dn), cols 16..31 = h[1:17]
  if (lane < 16) {
    int p = lane;
    int gp = tileBase + p;
    outSigma[gp] = expf(scrF[p * 32 + 0]);
    float dx = D[gp * 3 + 0], dy = D[gp * 3 + 1], dz = D[gp * 3 + 2];
    float inv = 1.0f / sqrtf(dx * dx + dy * dy + dz * dz);
    float x = dx * inv, y = dy * inv, z = dz * inv;
    float x2 = x * x, y2 = y * y, z2 = z * z;
    float xy = x * y, xz = x * z, yz = y * z;
    float sh[16];
    sh[0]  = 0.28209479177387814f;
    sh[1]  = -0.48860251190291987f * y;
    sh[2]  =  0.48860251190291987f * z;
    sh[3]  = -0.48860251190291987f * x;
    sh[4]  =  1.0925484305920792f * xy;
    sh[5]  = -1.0925484305920792f * yz;
    sh[6]  =  0.94617469575756f * z2 - 0.31539156525252f;
    sh[7]  = -1.0925484305920792f * xz;
    sh[8]  =  0.5462742152960396f * (x2 - y2);
    sh[9]  =  0.5900435899266435f * y * (-3.0f * x2 + y2);
    sh[10] =  2.890611442640554f * xy * z;
    sh[11] =  0.4570457994644657f * y * (1.0f - 5.0f * z2);
    sh[12] =  0.3731763325901154f * z * (5.0f * z2 - 3.0f);
    sh[13] =  0.4570457994644657f * x * (1.0f - 5.0f * z2);
    sh[14] =  1.445305721320277f * z * (x2 - y2);
    sh[15] =  0.5900435899266435f * x * (-x2 + 3.0f * y2);
#pragma unroll
    for (int j = 0; j < 16; ++j) bufA[p * 32 + j] = (_Float16)sh[j];
  } else {
    int p = lane - 16;
#pragma unroll
    for (int j = 0; j < 16; ++j)
      bufA[p * 32 + 16 + j] = (_Float16)scrF[p * 32 + 1 + j];
  }
  __syncthreads();

  // ---------------- 5. rgb MLP: 32->64 relu, 64->64 relu, 64->3, sigmoid ----
  mlpLayer<32, 64, true>(bufA, smemH + W_RGB0, bufB, lane);
  __syncthreads();
  mlpLayer<64, 64, true>(bufB, smemH + W_RGB1, bufC, lane);
  __syncthreads();
  mlpLayerF32<64, 16>(bufC, smemH + W_RGB2, scrF, lane);
  __syncthreads();
  if (lane < 16) {
    int p = lane;
    int gp = tileBase + p;
#pragma unroll
    for (int c = 0; c < 3; ++c) {
      float v = scrF[p * 16 + c];
      outRgb[gp * 3 + c] = 1.0f / (1.0f + expf(-v));
    }
  }
}

__global__ void init_load_kernel(float* out) {
  if (threadIdx.x < 2) out[6 * N_PTS + threadIdx.x] = 0.0f;
}

extern "C" void kernel_launch(void* const* d_in, const int* in_sizes, int n_in,
                              void* d_out, int out_size, void* d_ws, size_t ws_size,
                              hipStream_t stream) {
  const float* X     = (const float*)d_in[0];
  const float* D     = (const float*)d_in[1];
  const float* HT    = (const float*)d_in[2];
  const float* gw0   = (const float*)d_in[3];
  const float* gw1   = (const float*)d_in[4];
  const float* gw2   = (const float*)d_in[5];
  const float* ew0   = (const float*)d_in[6];
  const float* ew1   = (const float*)d_in[7];
  const float* ew2   = (const float*)d_in[8];
  const float* geow0 = (const float*)d_in[9];
  const float* geow1 = (const float*)d_in[10];
  const float* rw0   = (const float*)d_in[11];
  const float* rw1   = (const float*)d_in[12];
  const float* rw2   = (const float*)d_in[13];
  float* out = (float*)d_out;

  init_load_kernel<<<1, 32, 0, stream>>>(out);
  size_t smemBytes = (size_t)SMEM_HALFS * sizeof(_Float16);
  ngp_moe_kernel<<<NBLK, TPB, smemBytes, stream>>>(
      X, D, HT, gw0, gw1, gw2, ew0, ew1, ew2, geow0, geow1, rw0, rw1, rw2, out);
}
